// FEDformerEncoder_33277406609977
// MI455X (gfx1250) — matrix operand order repaired and checked by reading, not compiled, not run on hardware
//
#include <hip/hip_runtime.h>
#include <hip/hip_bf16.h>

// ---------------------------------------------------------------------------
// Types for CDNA5 WMMA (gfx1250, wave32)
// ---------------------------------------------------------------------------
typedef __attribute__((ext_vector_type(16))) __bf16 bf16x16;
typedef __attribute__((ext_vector_type(8)))  float  f32x8;
typedef __attribute__((ext_vector_type(4)))  int    v4i;

#define AS1 __attribute__((address_space(1)))
#define AS3 __attribute__((address_space(3)))

#if __has_builtin(__builtin_amdgcn_global_load_async_to_lds_b128)
#define HAVE_ASYNC_LDS 1
#else
#define HAVE_ASYNC_LDS 0
#endif

__device__ __forceinline__ void wait_async_zero() {
#if __has_builtin(__builtin_amdgcn_s_wait_asynccnt)
    __builtin_amdgcn_s_wait_asynccnt(0);
#else
    asm volatile("s_wait_asynccnt 0x0" ::: "memory");
#endif
}

union Frag16 {
    uint4   q[2];   // 32 bytes = 8 VGPRs
    bf16x16 v;
};

// round-to-nearest-even f32 -> bf16, packed pair
__device__ __forceinline__ unsigned pack_bf16x2(float a, float b) {
    unsigned ua = __float_as_uint(a);
    unsigned ub = __float_as_uint(b);
    ua = ua + 0x7FFFu + ((ua >> 16) & 1u);
    ub = ub + 0x7FFFu + ((ub >> 16) & 1u);
    return (ua >> 16) | (ub & 0xFFFF0000u);
}

// ---------------------------------------------------------------------------
// Kernel 0: elementwise f32 -> packed bf16 (weight conversion, once/layer)
// ---------------------------------------------------------------------------
__global__ __launch_bounds__(256) void f32_to_bf16_kernel(
        const float* __restrict__ src, unsigned short* __restrict__ dst, int n4)
{
    const int i = blockIdx.x * 256 + threadIdx.x;
    if (i < n4) {
        const float4 v = ((const float4*)src)[i];
        uint2 p;
        p.x = pack_bf16x2(v.x, v.y);
        p.y = pack_bf16x2(v.z, v.w);
        ((uint2*)dst)[i] = p;
    }
}

// ---------------------------------------------------------------------------
// Kernel 1: MoE conv. mean_e(conv(x, w_e) + b_e) == conv(x, mean_e w_e) + mean_e b_e
// One block (256 thr) per row of 2048. K=25 taps, pad=12. Output packed bf16.
// ---------------------------------------------------------------------------
#define CONV_K 25
#define CONV_PAD 12
#define WDIM 2048

__global__ __launch_bounds__(256) void conv_moe_kernel(
        const float* __restrict__ xin,        // [4096, 2048] f32
        const float* __restrict__ cw,         // [8, 25] (this layer)
        const float* __restrict__ cb,         // [8]
        unsigned short* __restrict__ out_bf)  // [4096, 2048] bf16
{
    __shared__ float tile[WDIM + CONV_K - 1];
    __shared__ float wf[CONV_K];
    __shared__ float bavg;

    const int row = blockIdx.x;
    const int tid = threadIdx.x;

    if (tid < CONV_K) {
        float s = 0.f;
        #pragma unroll
        for (int e = 0; e < 8; ++e) s += cw[e * CONV_K + tid];
        wf[tid] = s * 0.125f;
    }
    if (tid == CONV_K) {
        float s = 0.f;
        #pragma unroll
        for (int e = 0; e < 8; ++e) s += cb[e];
        bavg = s * 0.125f;
    }

    const float* xr = xin + (size_t)row * WDIM;
    for (int i = tid; i < WDIM + CONV_K - 1; i += 256) {
        int g = i - CONV_PAD;
        tile[i] = (g >= 0 && g < WDIM) ? xr[g] : 0.f;
    }
    __syncthreads();

    unsigned* orow = (unsigned*)(out_bf + (size_t)row * WDIM);
    for (int w2 = tid; w2 < WDIM / 2; w2 += 256) {
        const int w = 2 * w2;
        float s0 = bavg, s1 = bavg;
        #pragma unroll
        for (int t = 0; t < CONV_K; ++t) {
            s0 += tile[w + t] * wf[t];
            s1 += tile[w + 1 + t] * wf[t];
        }
        orow[w2] = pack_bf16x2(s0, s1);
    }
}

// ---------------------------------------------------------------------------
// Kernel 2: GEMM  C[M,N] = A[M,K] * B[N,K]^T + bias[N]   (bf16 in, f32 out)
// Block tile 128x128, BK=32, 256 threads = 8 waves (4M x 2N wave grid,
// each wave 32x64 -> 2x4 v_wmma_f32_16x16x32_bf16).
//
// Software pipeline with one LDS buffer:
//   frag ds_loads -> barrier -> issue ASYNC global->LDS loads of tile k+1
//   -> 8 WMMAs on in-register fragments -> s_wait_asynccnt 0 -> barrier.
// Primary path uses GLOBAL_LOAD_ASYNC_TO_LDS_B128 (ASYNCcnt); fallback stages
// through 4 distinct uint4 temps so loads batch ahead of the ds_stores.
// ---------------------------------------------------------------------------
#define BM 128
#define BN 128
#define BK 32
#define LDT 36   // dword stride per LDS tile row (16 data dwords + pad, 144B)

struct Stage { uint4 a0, a1, b0, b1; };

__global__ __launch_bounds__(256, 2) void gemm_bf16_wmma_kernel(
        const unsigned short* __restrict__ A,     // [M, K] bf16
        const unsigned short* __restrict__ B,     // [N, K] bf16
        const float* __restrict__ bias,           // [N]
        float* __restrict__ C,                    // [M, N] f32
        int M, int N, int K)
{
    __shared__ unsigned As[BM * LDT];
    __shared__ unsigned Bs[BN * LDT];

    const int tid  = threadIdx.x;
    const int bm   = blockIdx.y * BM;
    const int bn   = blockIdx.x * BN;
    const int lane = tid & 31;
    const int wid  = tid >> 5;
    const int wm   = (wid & 3) * 32;   // wave M offset within block tile
    const int wn   = (wid >> 2) * 64;  // wave N offset within block tile
    const int half = lane >> 4;        // 0: lanes 0-15, 1: lanes 16-31
    const int ln   = lane & 15;

    // staging slots: 2 per matrix per thread; slot s -> row s>>2, 16B group s&3
    const int r_0 = tid >> 2;            // slot tid
    const int g_0 = tid & 3;
    const int r_1 = (tid + 256) >> 2;    // slot tid+256
    const int g_1 = (tid + 256) & 3;

    const unsigned short* gA0 = A + (size_t)(bm + r_0) * K + g_0 * 8;
    const unsigned short* gA1 = A + (size_t)(bm + r_1) * K + g_1 * 8;
    const unsigned short* gB0 = B + (size_t)(bn + r_0) * K + g_0 * 8;
    const unsigned short* gB1 = B + (size_t)(bn + r_1) * K + g_1 * 8;
    unsigned* lA0 = &As[r_0 * LDT + g_0 * 4];
    unsigned* lA1 = &As[r_1 * LDT + g_1 * 4];
    unsigned* lB0 = &Bs[r_0 * LDT + g_0 * 4];
    unsigned* lB1 = &Bs[r_1 * LDT + g_1 * 4];

    f32x8 acc[2][4];
    #pragma unroll
    for (int i = 0; i < 2; ++i)
        #pragma unroll
        for (int j = 0; j < 4; ++j)
            acc[i][j] = (f32x8){0.f,0.f,0.f,0.f,0.f,0.f,0.f,0.f};

    Stage st;

    // ---- issue staging for K offset k0 ----
    auto stage_issue = [&](int k0, Stage& s) {
#if HAVE_ASYNC_LDS
        __builtin_amdgcn_global_load_async_to_lds_b128(
            (AS1 v4i*)(void*)(gA0 + k0), (AS3 v4i*)(void*)lA0, 0, 0);
        __builtin_amdgcn_global_load_async_to_lds_b128(
            (AS1 v4i*)(void*)(gA1 + k0), (AS3 v4i*)(void*)lA1, 0, 0);
        __builtin_amdgcn_global_load_async_to_lds_b128(
            (AS1 v4i*)(void*)(gB0 + k0), (AS3 v4i*)(void*)lB0, 0, 0);
        __builtin_amdgcn_global_load_async_to_lds_b128(
            (AS1 v4i*)(void*)(gB1 + k0), (AS3 v4i*)(void*)lB1, 0, 0);
        (void)s;
#else
        s.a0 = *(const uint4*)(gA0 + k0);
        s.a1 = *(const uint4*)(gA1 + k0);
        s.b0 = *(const uint4*)(gB0 + k0);
        s.b1 = *(const uint4*)(gB1 + k0);
#endif
    };
    // ---- make staged tile visible in LDS (this wave's part) ----
    auto stage_commit = [&](Stage& s) {
#if HAVE_ASYNC_LDS
        wait_async_zero();
        (void)s;
#else
        *(uint4*)lA0 = s.a0;
        *(uint4*)lA1 = s.a1;
        *(uint4*)lB0 = s.b0;
        *(uint4*)lB1 = s.b1;
#endif
    };

    // prologue: stage tile 0
    stage_issue(0, st);
    stage_commit(st);
    __syncthreads();

    for (int k0 = 0; k0 < K; k0 += BK) {
        // ---- fragments in contiguous k order (same for A and B) ----
        // A 16x32: lanes 0-15 (M=ln) K 0..15; lanes 16-31 K 16..31
        Frag16 a[2];
        #pragma unroll
        for (int i = 0; i < 2; ++i) {
            const int rr = wm + 16 * i + ln;
            const uint4* p = (const uint4*)&As[rr * LDT + half * 8];
            a[i].q[0] = p[0];   // dwords +0..3
            a[i].q[1] = p[1];   // dwords +4..7
        }
        // B 32x16: lanes 0-15 (N=ln) K 0..15; lanes 16-31 K 16..31
        Frag16 b[4];
        #pragma unroll
        for (int j = 0; j < 4; ++j) {
            const int rr = wn + 16 * j + ln;
            const uint4* p = (const uint4*)&Bs[rr * LDT + half * 8];
            b[j].q[0] = p[0];
            b[j].q[1] = p[1];
        }

        __syncthreads();                 // all waves done reading this LDS tile

        const bool more = (k0 + BK) < K;
        if (more) stage_issue(k0 + BK, st);   // overlap next-tile fill with WMMA

        #pragma unroll
        for (int i = 0; i < 2; ++i)
            #pragma unroll
            for (int j = 0; j < 4; ++j)
                acc[i][j] = __builtin_amdgcn_wmma_f32_16x16x32_bf16(
                    false, a[i].v, false, b[j].v, (short)0, acc[i][j], false, false);

        if (more) {
            stage_commit(st);
            __syncthreads();             // next tile fully resident
        }
    }

    // ---- epilogue: lanes 0-15: N=ln, M=r; lanes 16-31: M=8+r
    #pragma unroll
    for (int j = 0; j < 4; ++j) {
        const int col = bn + wn + 16 * j + ln;
        const float bb = bias[col];
        #pragma unroll
        for (int i = 0; i < 2; ++i) {
            const int rowb = bm + wm + 16 * i + half * 8;
            #pragma unroll
            for (int r = 0; r < 8; ++r)
                C[(size_t)(rowb + r) * N + col] = acc[i][j][r] + bb;
        }
    }
}

// ---------------------------------------------------------------------------
// Kernel 3: FFT attention per row (P = 1024).
// out = real(ifft( conj(fft q) * fft k * fft v / sqrt(P) ))  -> packed bf16
// real(ifft(W)) == real(fft(conj(W))) / P  : 4 forward Stockham FFTs in LDS.
// ---------------------------------------------------------------------------
#define PDIM 1024

__device__ __forceinline__ void fft1024_fwd(float2* __restrict__ a,
                                            float2* __restrict__ b,
                                            int tid)
{
    float2* x = a;
    float2* y = b;
    #pragma unroll
    for (int s = 0; s < 10; ++s) {
        const int L = 1 << s;
        __syncthreads();
        #pragma unroll
        for (int it = 0; it < 2; ++it) {
            const int i = tid + it * 256;          // 512 butterflies
            const int kk = i & (L - 1);
            const int j  = ((i >> s) << (s + 1)) + kk;
            const float2 u = x[i];
            const float2 w = x[i + 512];
            const float ang = -3.14159265358979323846f * (float)kk / (float)L;
            float sn, cs;
            __sincosf(ang, &sn, &cs);
            const float tr = w.x * cs - w.y * sn;
            const float ti = w.x * sn + w.y * cs;
            y[j]     = make_float2(u.x + tr, u.y + ti);
            y[j + L] = make_float2(u.x - tr, u.y - ti);
        }
        float2* tmp = x; x = y; y = tmp;
    }
    __syncthreads();   // result back in 'a' (10 swaps)
}

__global__ __launch_bounds__(256) void fft_attn_kernel(
        const float* __restrict__ q,              // [4096, 1024] f32
        const float* __restrict__ k,
        const float* __restrict__ v,
        unsigned short* __restrict__ outw_bf)     // [4096, 1024] bf16
{
    __shared__ float2 s0[PDIM];
    __shared__ float2 s1[PDIM];
    __shared__ float2 Qf[PDIM];
    __shared__ float2 Kf[PDIM];

    const int row = blockIdx.x;
    const int tid = threadIdx.x;
    const size_t base = (size_t)row * PDIM;

    // Q
    for (int i = tid; i < PDIM; i += 256) s0[i] = make_float2(q[base + i], 0.f);
    fft1024_fwd(s0, s1, tid);
    for (int i = tid; i < PDIM; i += 256) Qf[i] = s0[i];
    // K
    for (int i = tid; i < PDIM; i += 256) s0[i] = make_float2(k[base + i], 0.f);
    fft1024_fwd(s0, s1, tid);
    for (int i = tid; i < PDIM; i += 256) Kf[i] = s0[i];
    // V
    for (int i = tid; i < PDIM; i += 256) s0[i] = make_float2(v[base + i], 0.f);
    fft1024_fwd(s0, s1, tid);

    // W = conj(Qf)*Kf*Vf / 32 ; store conj(W) for the inverse-via-forward trick
    const float inv_scale = 1.0f / 32.0f;   // 1/sqrt(1024)
    for (int i = tid; i < PDIM; i += 256) {
        const float2 Q = Qf[i], K2 = Kf[i], V2 = s0[i];
        const float ar = Q.x * K2.x + Q.y * K2.y;   // conj(Q)*K
        const float ai = Q.x * K2.y - Q.y * K2.x;
        const float wr = (ar * V2.x - ai * V2.y) * inv_scale;
        const float wi = (ar * V2.y + ai * V2.x) * inv_scale;
        s0[i] = make_float2(wr, -wi);               // conj(W)
    }
    fft1024_fwd(s0, s1, tid);

    const float invN = 1.0f / (float)PDIM;
    unsigned* op = (unsigned*)(outw_bf + base);
    for (int i2 = tid; i2 < PDIM / 2; i2 += 256)
        op[i2] = pack_bf16x2(s0[2 * i2].x * invN, s0[2 * i2 + 1].x * invN);
}

// ---------------------------------------------------------------------------
// Host-side orchestration
// ---------------------------------------------------------------------------
extern "C" void kernel_launch(void* const* d_in, const int* in_sizes, int n_in,
                              void* d_out, int out_size, void* d_ws, size_t ws_size,
                              hipStream_t stream) {
    (void)in_sizes; (void)n_in; (void)out_size; (void)ws_size;

    const float* x   = (const float*)d_in[0];   // [32,128,2048]
    const float* cw  = (const float*)d_in[1];   // [2,8,1,25]
    const float* cb  = (const float*)d_in[2];   // [2,8]
    const float* wq  = (const float*)d_in[3];   // [2,1024,2048]
    const float* bq  = (const float*)d_in[4];   // [2,1024]
    const float* wk  = (const float*)d_in[5];
    const float* bk  = (const float*)d_in[6];
    const float* wvp = (const float*)d_in[7];
    const float* bv  = (const float*)d_in[8];
    const float* wo  = (const float*)d_in[9];   // [2,2048,1024]
    const float* bo  = (const float*)d_in[10];  // [2,2048]
    float* out = (float*)d_out;

    char* wsb = (char*)d_ws;
    const size_t MB = (size_t)1 << 20;
    unsigned short* u_bf   = (unsigned short*)(wsb + 0);        // [4096,2048] bf16 (16 MB)
    float*          qb     = (float*)(wsb + 16 * MB);           // [4096,1024] f32  (16 MB)
    float*          kb     = (float*)(wsb + 32 * MB);
    float*          vb     = (float*)(wsb + 48 * MB);
    unsigned short* wvr_bf = (unsigned short*)(wsb + 64 * MB);  // [4096,1024] bf16 ( 8 MB)
    float*          xcur   = (float*)(wsb + 72 * MB);           // [4096,2048] f32  (32 MB)
    unsigned short* wq_bf  = (unsigned short*)(wsb + 104 * MB); // [1024,2048] bf16 ( 4 MB)
    unsigned short* wk_bf  = (unsigned short*)(wsb + 108 * MB);
    unsigned short* wv_bf  = (unsigned short*)(wsb + 112 * MB);
    unsigned short* wo_bf  = (unsigned short*)(wsb + 116 * MB); // [2048,1024] bf16 ( 4 MB)

    const int M = 32 * 128;                     // 4096 rows
    const int NW4 = (1024 * 2048) / 4;          // float4 count per weight matrix
    const int CVT_GRID = NW4 / 256;

    for (int l = 0; l < 2; ++l) {
        const float* xin = (l == 0) ? x : xcur;

        // per-layer weight conversion (f32 -> bf16), trivial vs. GEMM cost
        f32_to_bf16_kernel<<<CVT_GRID, 256, 0, stream>>>(wq  + (size_t)l * 1024 * 2048, wq_bf, NW4);
        f32_to_bf16_kernel<<<CVT_GRID, 256, 0, stream>>>(wk  + (size_t)l * 1024 * 2048, wk_bf, NW4);
        f32_to_bf16_kernel<<<CVT_GRID, 256, 0, stream>>>(wvp + (size_t)l * 1024 * 2048, wv_bf, NW4);
        f32_to_bf16_kernel<<<CVT_GRID, 256, 0, stream>>>(wo  + (size_t)l * 2048 * 1024, wo_bf, NW4);

        conv_moe_kernel<<<M, 256, 0, stream>>>(
            xin, cw + (size_t)l * 8 * 25, cb + (size_t)l * 8, u_bf);

        const dim3 gq(1024 / BN, M / BM);   // q/k/v: N=1024, K=2048
        gemm_bf16_wmma_kernel<<<gq, 256, 0, stream>>>(u_bf, wq_bf, bq + (size_t)l * 1024, qb, M, 1024, 2048);
        gemm_bf16_wmma_kernel<<<gq, 256, 0, stream>>>(u_bf, wk_bf, bk + (size_t)l * 1024, kb, M, 1024, 2048);
        gemm_bf16_wmma_kernel<<<gq, 256, 0, stream>>>(u_bf, wv_bf, bv + (size_t)l * 1024, vb, M, 1024, 2048);

        fft_attn_kernel<<<M, 256, 0, stream>>>(qb, kb, vb, wvr_bf);

        float* dst = (l == 1) ? out : xcur;
        const dim3 go(2048 / BN, M / BM);   // out-proj: N=2048, K=1024
        gemm_bf16_wmma_kernel<<<go, 256, 0, stream>>>(wvr_bf, wo_bf, bo + (size_t)l * 2048, dst, M, 2048, 1024);
    }
}
